// AttentiveTransformer_64957085385018
// MI455X (gfx1250) — matrix operand order repaired and checked by reading, not compiled, not run on hardware
//
#include <hip/hip_runtime.h>

typedef __attribute__((ext_vector_type(16))) __bf16 v16bf;
typedef __attribute__((ext_vector_type(8)))  __bf16 v8bf;
typedef __attribute__((ext_vector_type(8)))  float  v8f;

#define D_IN  512
#define D_OUT 256
#define ROWS  32          // rows per block
#define KC    32          // K chunk

__device__ __forceinline__ __bf16 f2bf(float f) {
  unsigned u = __builtin_bit_cast(unsigned, f);
  unsigned short h = (unsigned short)(u >> 16);   // truncate: residual stays exact
  return __builtin_bit_cast(__bf16, h);
}
__device__ __forceinline__ float bf2f(__bf16 b) {
  unsigned u = ((unsigned)__builtin_bit_cast(unsigned short, b)) << 16;
  return __builtin_bit_cast(float, u);
}
__device__ __forceinline__ unsigned pack2(__bf16 a, __bf16 b) {
  return (unsigned)__builtin_bit_cast(unsigned short, a) |
         ((unsigned)__builtin_bit_cast(unsigned short, b) << 16);
}

// Async copy 16B global -> LDS (CDNA5 GLOBAL_LOAD_ASYNC_TO_LDS_B128, ASYNCcnt).
__device__ __forceinline__ void async_cp16(uint32_t lds_off, const void* gptr) {
  asm volatile("global_load_async_to_lds_b128 %0, %1, off"
               :: "v"(lds_off), "v"(gptr) : "memory");
}
__device__ __forceinline__ void async_wait0() {
  asm volatile("s_wait_asynccnt 0x0" ::: "memory");
}

// ---------------------------------------------------------------------------
// Prep kernel: split W (f32 [512][256]) into bf16 hi/lo, stored chunk-major:
//   wt[chunk][n][k%32], chunk = k/32  -> each 16KB K-chunk is contiguous.
// ---------------------------------------------------------------------------
__global__ __launch_bounds__(256)
void w_split_kernel(const float* __restrict__ W,
                    __bf16* __restrict__ wt_hi, __bf16* __restrict__ wt_lo) {
  const int e = blockIdx.x * 256 + threadIdx.x;   // over 512*256
  const int k = e >> 8;
  const int n = e & 255;
  const float w = W[(size_t)k * D_OUT + n];
  const __bf16 h = f2bf(w);
  const __bf16 l = f2bf(w - bf2f(h));
  const size_t idx = (((size_t)(k >> 5) * D_OUT) + n) * KC + (k & 31);
  wt_hi[idx] = h;
  wt_lo[idx] = l;
}

// LDS layout (bytes):
//   [0     .. 2048 )  A_hi [32][32] bf16
//   [2048  .. 4096 )  A_lo [32][32] bf16
//   [4096  .. 20480)  Bh   [256][32] bf16 (pre-split W chunk, hi)
//   [20480 .. 36864)  Bl   [256][32] bf16 (lo)
// Phase 2 aliases [0 .. 32768) as zbuf[32][256] f32.
__global__ __launch_bounds__(256)
void fused_attentive_kernel(const float* __restrict__ X,
                            const float* __restrict__ P,
                            const __bf16* __restrict__ wt_hi,
                            const __bf16* __restrict__ wt_lo,
                            const float* __restrict__ bns,
                            const float* __restrict__ bnb,
                            const float* __restrict__ bnm,
                            const float* __restrict__ bnv,
                            float* __restrict__ out) {
  __shared__ __align__(64) char smem[36864];
  unsigned* A_hi_w = (unsigned*)(smem);            // packed bf16 pairs
  unsigned* A_lo_w = (unsigned*)(smem + 2048);
  __bf16*   A_hi   = (__bf16*)(smem);
  __bf16*   A_lo   = (__bf16*)(smem + 2048);
  __bf16*   Bh     = (__bf16*)(smem + 4096);
  __bf16*   Bl     = (__bf16*)(smem + 20480);
  const uint32_t lds0 = (uint32_t)(uintptr_t)smem;  // LDS byte addr = low 32 bits

  const int tid  = threadIdx.x;
  const int wave = tid >> 5;
  const int lane = tid & 31;
  const int row0 = blockIdx.x * ROWS;

  // wave -> 16x64 output slab
  const int mt = wave & 1;           // M-tile (0/1)
  const int nb = (wave >> 1) << 6;   // N base (0..192)

  v8f acc[4] = {};

  for (int kc = 0; kc < D_IN; kc += KC) {
    // ---- async-stage pre-split W chunk: 2 x 16KB contiguous (64B/thread) ----
    {
      const size_t cbase = ((size_t)(kc >> 5) * D_OUT) * KC;  // chunk-major
      const __bf16* gh = wt_hi + cbase + (size_t)tid * KC;    // one n-row: 32 bf16
      const __bf16* gl = wt_lo + cbase + (size_t)tid * KC;
      const uint32_t lh = lds0 + 4096  + (uint32_t)tid * 64;
      const uint32_t ll = lds0 + 20480 + (uint32_t)tid * 64;
#pragma unroll
      for (int j = 0; j < 4; ++j) {
        async_cp16(lh + j * 16, gh + j * 8);
        async_cp16(ll + j * 16, gl + j * 8);
      }
    }
    // ---- stage X chunk: load f32, split once, packed b32 LDS stores ----
    {
      const int m  = tid >> 3;
      const int k4 = (tid & 7) << 2;
      const float4 xv = *reinterpret_cast<const float4*>(
          X + (size_t)(row0 + m) * D_IN + kc + k4);
      const float xs[4] = {xv.x, xv.y, xv.z, xv.w};
      __bf16 h[4], l[4];
#pragma unroll
      for (int j = 0; j < 4; ++j) {
        h[j] = f2bf(xs[j]);
        l[j] = f2bf(xs[j] - bf2f(h[j]));
      }
      const int widx = m * 16 + (k4 >> 1);
      A_hi_w[widx]     = pack2(h[0], h[1]);
      A_hi_w[widx + 1] = pack2(h[2], h[3]);
      A_lo_w[widx]     = pack2(l[0], l[1]);
      A_lo_w[widx + 1] = pack2(l[2], l[3]);
    }
    async_wait0();
    __syncthreads();

    // ---- A fragments: pure 16B LDS loads, no conversion.
    // A layout (matches B): lane m = lane%16, K: b+{0..15}, b=(lane>>4)*16
    const int am = mt * 16 + (lane & 15);
    const int ab = (lane >> 4) << 4;
    const v8bf a0h = *(const v8bf*)(A_hi + am * KC + ab);
    const v8bf a1h = *(const v8bf*)(A_hi + am * KC + ab + 8);
    const v8bf a0l = *(const v8bf*)(A_lo + am * KC + ab);
    const v8bf a1l = *(const v8bf*)(A_lo + am * KC + ab + 8);
    const v16bf ah = __builtin_shufflevector(a0h, a1h,
        0,1,2,3,4,5,6,7,8,9,10,11,12,13,14,15);
    const v16bf al = __builtin_shufflevector(a0l, a1l,
        0,1,2,3,4,5,6,7,8,9,10,11,12,13,14,15);

    // ---- B fragments + 3-pass split-precision WMMA per N-tile ----
    const int bncol = lane & 15;
    const int bk    = (lane >> 4) << 4;            // K half: 0 or 16
#pragma unroll
    for (int t = 0; t < 4; ++t) {
      const int nrow = (nb + t * 16 + bncol) * KC + bk;
      const v16bf bh = *(const v16bf*)(Bh + nrow);
      const v16bf bl = *(const v16bf*)(Bl + nrow);
      acc[t] = __builtin_amdgcn_wmma_f32_16x16x32_bf16(
          false, ah, false, bh, (short)0, acc[t], false, false);
      acc[t] = __builtin_amdgcn_wmma_f32_16x16x32_bf16(
          false, ah, false, bl, (short)0, acc[t], false, false);
      acc[t] = __builtin_amdgcn_wmma_f32_16x16x32_bf16(
          false, al, false, bh, (short)0, acc[t], false, false);
    }
    __syncthreads();
  }

  // ---- epilogue: BN affine + prior scale -> zbuf (aliases staging LDS) ----
  float* zbuf = (float*)smem;
#pragma unroll
  for (int t = 0; t < 4; ++t) {
    const int gn   = nb + t * 16 + (lane & 15);
    const float mu = bnm[gn];
    const float iv = bns[gn] * rsqrtf(bnv[gn] + 1e-5f);
    const float be = bnb[gn];
#pragma unroll
    for (int r = 0; r < 8; ++r) {
      // ISA 32-bit C/D layout: lanes0-15 -> M=r, lanes16-31 -> M=8+r
      const int m = mt * 16 + ((lane >> 4) << 3) + r;
      const size_t grow = (size_t)(row0 + m);
      float z = (acc[t][r] - mu) * iv + be;
      z *= P[grow * D_OUT + gn];
      zbuf[m * D_OUT + gn] = z;
    }
  }
  __syncthreads();

  // ---- sparsemax per row: Michelot fixed-point (exact simplex projection) ----
#pragma unroll
  for (int i = 0; i < 4; ++i) {
    const int m = (wave << 2) + i;
    const float4 z0 = *(const float4*)(zbuf + m * D_OUT + lane * 8);
    const float4 z1 = *(const float4*)(zbuf + m * D_OUT + lane * 8 + 4);
    float v[8] = {z0.x, z0.y, z0.z, z0.w, z1.x, z1.y, z1.z, z1.w};

    // tau_0 = (sum(v) - 1)/d ; then tau <- (sum_{v>tau} v - 1)/count_{v>tau}.
    // Active set shrinks monotonically; fixed point is the exact sparsemax tau.
    float s0 = 0.0f;
#pragma unroll
    for (int j = 0; j < 8; ++j) s0 += v[j];
#pragma unroll
    for (int off = 16; off > 0; off >>= 1) s0 += __shfl_xor(s0, off, 32);
    float tau = (s0 - 1.0f) * (1.0f / 256.0f);

#pragma unroll
    for (int it = 0; it < 8; ++it) {
      float s = 0.0f, c = 0.0f;
#pragma unroll
      for (int j = 0; j < 8; ++j) {
        const bool a = v[j] > tau;
        s += a ? v[j] : 0.0f;
        c += a ? 1.0f : 0.0f;
      }
#pragma unroll
      for (int off = 16; off > 0; off >>= 1) {
        s += __shfl_xor(s, off, 32);
        c += __shfl_xor(c, off, 32);
      }
      tau = (s - 1.0f) / c;   // c >= 1 (max element always active)
    }

    const size_t grow = (size_t)(row0 + m);
    float4 o0, o1;
    o0.x = fmaxf(v[0] - tau, 0.0f); o0.y = fmaxf(v[1] - tau, 0.0f);
    o0.z = fmaxf(v[2] - tau, 0.0f); o0.w = fmaxf(v[3] - tau, 0.0f);
    o1.x = fmaxf(v[4] - tau, 0.0f); o1.y = fmaxf(v[5] - tau, 0.0f);
    o1.z = fmaxf(v[6] - tau, 0.0f); o1.w = fmaxf(v[7] - tau, 0.0f);
    *reinterpret_cast<float4*>(out + grow * D_OUT + lane * 8)     = o0;
    *reinterpret_cast<float4*>(out + grow * D_OUT + lane * 8 + 4) = o1;
  }
}

extern "C" void kernel_launch(void* const* d_in, const int* in_sizes, int n_in,
                              void* d_out, int out_size, void* d_ws, size_t ws_size,
                              hipStream_t stream) {
  (void)n_in; (void)out_size; (void)ws_size;
  const float* X   = (const float*)d_in[0];
  const float* P   = (const float*)d_in[1];
  const float* W   = (const float*)d_in[2];
  const float* bns = (const float*)d_in[3];
  const float* bnb = (const float*)d_in[4];
  const float* bnm = (const float*)d_in[5];
  const float* bnv = (const float*)d_in[6];
  float* out = (float*)d_out;

  // Workspace: wt_hi [512*256] bf16 (256KB) + wt_lo (256KB)
  __bf16* wt_hi = (__bf16*)d_ws;
  __bf16* wt_lo = (__bf16*)((char*)d_ws + D_IN * D_OUT * sizeof(__bf16));

  w_split_kernel<<<dim3(D_IN * D_OUT / 256), dim3(256), 0, stream>>>(W, wt_hi, wt_lo);

  const int B = in_sizes[0] / D_IN;      // 131072
  fused_attentive_kernel<<<dim3(B / ROWS), dim3(256), 0, stream>>>(
      X, P, wt_hi, wt_lo, bns, bnb, bnm, bnv, out);
}
